// EncoderLayer_15865609191796
// MI455X (gfx1250) — matrix operand order repaired and checked by reading, not compiled, not run on hardware
//
#include <hip/hip_runtime.h>

// ---------------------------------------------------------------------------
// CDNA5 (gfx1250) transformer encoder layer, bf16 WMMA + fp32 accumulate.
// B=2, S=2048, D=1024, H=16, DK=64, FF=4096.
// B-side operands stored K-major: each lane's WMMA B fragment is a contiguous
// 32-byte run -> two 128-bit loads. GEMM: wave tile 32x64 (each B fragment
// feeds two WMMAs), double-buffered LDS B tile with register staging.
// ---------------------------------------------------------------------------

typedef __attribute__((ext_vector_type(16))) __bf16 v16bf;
typedef __attribute__((ext_vector_type(8)))  float  v8f;
typedef __attribute__((ext_vector_type(8)))  unsigned int v8u;
typedef __attribute__((ext_vector_type(4)))  unsigned int v4u;

union Frag { v16bf h; v8u u; };

__device__ __forceinline__ v8f wmma_bf16(v16bf a, v16bf b, v8f c) {
  // D = A(16x32 bf16) * B(32x16 bf16) + C(16x16 f32)
  return __builtin_amdgcn_wmma_f32_16x16x32_bf16(
      /*neg_a=*/false, a, /*neg_b=*/false, b,
      /*c_mod=*/(short)0, c, /*reuse_a=*/false, /*reuse_b=*/false);
}

// A fragment 16x32 bf16 from row-major A (leading dim lda), k offset k0.
// Lane<16 row=lane, hk=lane>>4; dword v holds K = hk*16 + 2v, 2v+1
// -> two contiguous 16B chunks at elem offsets {0, 8} from row*lda + k0 + hk*16.
__device__ __forceinline__ v16bf load_a_frag(const __bf16* A, int lda, int k0, int lane) {
  const __bf16* base = A + (size_t)(lane & 15) * lda + k0 + (lane >> 4) * 16;
  const v4u lo = *(const v4u*)(base);
  const v4u hi = *(const v4u*)(base + 8);
  Frag f;
#pragma unroll
  for (int i = 0; i < 4; ++i) { f.u[i] = lo[i]; f.u[i + 4] = hi[i]; }
  return f.h;
}

// B fragment 32x16 where B[k][n] = Bt[n][k] (Bt row-major [N, ld]).
// Lane: n = n0 + (lane&15), kh = lane>>4; dword v holds K = kh*16 + 2v, 2v+1
// -> one contiguous 32B run at Bt + n*ld + k0 + kh*16.
__device__ __forceinline__ v16bf load_bT_frag(const __bf16* Bt, int ld, int k0, int n0, int lane) {
  const __bf16* base = Bt + (size_t)(n0 + (lane & 15)) * ld + k0 + (lane >> 4) * 16;
  const v4u lo = *(const v4u*)(base);
  const v4u hi = *(const v4u*)(base + 8);
  Frag f;
#pragma unroll
  for (int i = 0; i < 4; ++i) { f.u[i] = lo[i]; f.u[i + 4] = hi[i]; }
  return f.h;
}

// ---------------------------------------------------------------------------
// fp32 -> bf16 convert (row-major copy)
// ---------------------------------------------------------------------------
__global__ void k_f32_to_bf16(const float* __restrict__ in, __bf16* __restrict__ out, int n) {
  int i = blockIdx.x * 256 + threadIdx.x;
  if (i < n) out[i] = (__bf16)in[i];
}

// fp32 [K,N] -> bf16 transposed [N,K]  (out[n*K+k] = in[k*N+n]); write-coalesced
__global__ void k_f32_to_bf16_T(const float* __restrict__ in, __bf16* __restrict__ out,
                                int K, int N) {
  int i = blockIdx.x * 256 + threadIdx.x;
  if (i >= K * N) return;
  const int n = i / K, k = i - n * K;
  out[i] = (__bf16)in[(size_t)k * N + n];
}

// ---------------------------------------------------------------------------
// GEMM: C[M,N] = A[M,K](bf16,row-major) @ B (given as Bt[N,K] bf16) + bias[N]
// mode 0: store fp32 to Cf.  mode 1: store bf16(relu(.)) to Cb.
// Block = 256 threads (8 waves). Block tile 256x64; wave tile 32x64.
// Double-buffered 64x32 K-major B tile in LDS; global prefetch distance 2
// iterations via a staging register; one barrier per k-step.
// ---------------------------------------------------------------------------
__global__ void __launch_bounds__(256) k_gemm_bf16(
    const __bf16* __restrict__ A, const __bf16* __restrict__ Bt,
    const float* __restrict__ bias, float* __restrict__ Cf,
    __bf16* __restrict__ Cb, int M, int N, int K, int relu_bf16_out) {
  __shared__ __attribute__((aligned(16))) __bf16 Bs[2][64 * 32];  // [buf][n][k0..31]
  const int lane = threadIdx.x & 31;
  const int wave = threadIdx.x >> 5;
  const int col0 = blockIdx.x * 64;
  const int row0 = blockIdx.y * 256 + wave * 32;
  const __bf16* Arow0 = A + (size_t)row0 * K;
  const __bf16* Arow1 = Arow0 + (size_t)16 * K;

  // staging coords: 256 threads x 16B = 4KB tile
  const int sn = threadIdx.x >> 2;          // 0..63   (n_local)
  const int sp = (threadIdx.x & 3) * 8;     // 0,8,16,24 (k elem offset)
  const __bf16* Bt_base = Bt + (size_t)(col0 + sn) * K + sp;
  const int soff = sn * 32 + sp;

  const int T = K >> 5;                     // number of 32-wide k tiles
  // prologue: tile 0 straight to LDS, tile 1 into the staging register
  v4u stg = *(const v4u*)(Bt_base);
  *(v4u*)(&Bs[0][soff]) = stg;
  if (T > 1) stg = *(const v4u*)(Bt_base + 32);

  v8f acc0[4] = {};
  v8f acc1[4] = {};
  for (int kt = 0; kt < T; ++kt) {
    const int k = kt << 5;
    __syncthreads();  // buf[kt&1] ready; prior reads of buf[(kt+1)&1] done
    if (kt + 1 < T) {
      *(v4u*)(&Bs[(kt + 1) & 1][soff]) = stg;                 // data prefetched 1 iter ago
      if (kt + 2 < T) stg = *(const v4u*)(Bt_base + (size_t)(kt + 2) * 32);
    }
    const __bf16* Bcur = Bs[kt & 1];
    const v16bf a0 = load_a_frag(Arow0, K, k, lane);
    const v16bf a1 = load_a_frag(Arow1, K, k, lane);
    // rotate B-fragment loads one tile ahead of the WMMA pair
    v16bf b = load_bT_frag(Bcur, 32, 0, 0, lane);
#pragma unroll
    for (int t = 0; t < 4; ++t) {
      const v16bf bn = (t < 3) ? load_bT_frag(Bcur, 32, 0, (t + 1) * 16, lane) : b;
      acc0[t] = wmma_bf16(a0, b, acc0[t]);
      acc1[t] = wmma_bf16(a1, b, acc1[t]);
      b = bn;
    }
  }

  const int nl = lane & 15, hi = lane >> 4;
#pragma unroll
  for (int t = 0; t < 4; ++t) {
    const int n = col0 + t * 16 + nl;
    const float bv = bias[n];
#pragma unroll
    for (int v = 0; v < 8; ++v) {
      const int m0 = row0 + hi * 8 + v;
      const int m1 = m0 + 16;
      const float val0 = acc0[t][v] + bv;
      const float val1 = acc1[t][v] + bv;
      if (relu_bf16_out) {
        Cb[(size_t)m0 * N + n] = (__bf16)fmaxf(val0, 0.0f);
        Cb[(size_t)m1 * N + n] = (__bf16)fmaxf(val1, 0.0f);
      } else {
        Cf[(size_t)m0 * N + n] = val0;
        Cf[(size_t)m1 * N + n] = val1;
      }
    }
  }
}

// ---------------------------------------------------------------------------
// Split heads: C[(b*S+s)*1024 + h*64+d] -> out[((b*16+h)*2048+s)*64+d] * scale
// ---------------------------------------------------------------------------
__global__ void k_split_heads(const float* __restrict__ C, __bf16* __restrict__ out,
                              float scale, int total) {
  const int i = blockIdx.x * 256 + threadIdx.x;
  if (i >= total) return;
  const int row = i >> 10;       // / 1024
  const int c   = i & 1023;
  const int h   = c >> 6, d = c & 63;
  const int b   = row >> 11;     // / 2048
  const int s   = row & 2047;
  out[(((size_t)(b * 16 + h) * 2048) + s) * 64 + d] = (__bf16)(C[i] * scale);
}

// V variant, stores head-transposed: out[((b*16+h)*64 + d)*2048 + s]
__global__ void k_split_heads_T(const float* __restrict__ C, __bf16* __restrict__ out,
                                int total) {
  const int i = blockIdx.x * 256 + threadIdx.x;
  if (i >= total) return;
  const int row = i >> 10;
  const int c   = i & 1023;
  const int h   = c >> 6, d = c & 63;
  const int b   = row >> 11;
  const int s   = row & 2047;
  out[(((size_t)(b * 16 + h) * 64) + d) * 2048 + s] = (__bf16)C[i];
}

// ---------------------------------------------------------------------------
// Flash attention. Grid: (S/64, B*H), block 128 (4 waves).
// Wave handles 16 q-rows; loops 32-key tiles with online softmax.
// Q pre-scaled by 1/sqrt(dk). K is [bh][s][64] (read K-major via load_bT_frag),
// V is stored transposed [bh][d][s]. ctx written bf16 in [B,S,H*64].
// ---------------------------------------------------------------------------
__global__ void __launch_bounds__(128) k_flash_attn(
    const __bf16* __restrict__ Q, const __bf16* __restrict__ Kh,
    const __bf16* __restrict__ Vt, __bf16* __restrict__ ctx, int S, int H) {
  __shared__ __attribute__((aligned(16))) __bf16 pbuf[4][16 * 32];
  const int lane = threadIdx.x & 31;
  const int wave = threadIdx.x >> 5;
  const int bh = blockIdx.y;
  const int b = bh / H, h = bh % H;
  const int q0 = blockIdx.x * 64 + wave * 16;

  const __bf16* Qp = Q  + (size_t)bh * S * 64;
  const __bf16* Kp = Kh + (size_t)bh * S * 64;
  const __bf16* Vp = Vt + (size_t)bh * 64 * S;   // [d][s]

  const v16bf qa0 = load_a_frag(Qp + (size_t)q0 * 64, 64, 0,  lane);
  const v16bf qa1 = load_a_frag(Qp + (size_t)q0 * 64, 64, 32, lane);

  float mrow[8], lrow[8];
#pragma unroll
  for (int v = 0; v < 8; ++v) { mrow[v] = -1e30f; lrow[v] = 0.0f; }
  v8f O[4] = {};

  __bf16* pb = pbuf[wave];
  const int nl = lane & 15, hi = lane >> 4;

  for (int j = 0; j < S; j += 32) {
    // scores tile 16x32 = two 16x16 accumulators, contraction over dk=64
    v8f s0 = {}, s1 = {};
    s0 = wmma_bf16(qa0, load_bT_frag(Kp, 64, 0,  j,      lane), s0);
    s0 = wmma_bf16(qa1, load_bT_frag(Kp, 64, 32, j,      lane), s0);
    s1 = wmma_bf16(qa0, load_bT_frag(Kp, 64, 0,  j + 16, lane), s1);
    s1 = wmma_bf16(qa1, load_bT_frag(Kp, 64, 32, j + 16, lane), s1);

    // online softmax: element v <-> row (hi*8+v); 16 lanes of same hi = cols
#pragma unroll
    for (int v = 0; v < 8; ++v) {
      float mx = fmaxf(s0[v], s1[v]);
#pragma unroll
      for (int msk = 1; msk < 16; msk <<= 1)
        mx = fmaxf(mx, __shfl_xor(mx, msk, 32));
      const float mn = fmaxf(mrow[v], mx);
      const float alpha = __expf(mrow[v] - mn);
      const float p0 = __expf(s0[v] - mn);
      const float p1 = __expf(s1[v] - mn);
      float rs = p0 + p1;
#pragma unroll
      for (int msk = 1; msk < 16; msk <<= 1)
        rs += __shfl_xor(rs, msk, 32);
      lrow[v] = lrow[v] * alpha + rs;
      mrow[v] = mn;
      O[0][v] *= alpha; O[1][v] *= alpha; O[2][v] *= alpha; O[3][v] *= alpha;
      // C-layout -> row-major 16x32 bf16 in per-wave LDS scratch (transpose)
      pb[(v + 8 * hi) * 32 + nl]      = (__bf16)p0;
      pb[(v + 8 * hi) * 32 + 16 + nl] = (__bf16)p1;
    }
    // same-wave LDS ops are in-order: read P back as an A-fragment (2x b128)
    v16bf pa = load_a_frag(pb, 32, 0, lane);
    // P(16x32) @ V(32x64): B[k][n] = V[j+k][d=n] = Vp[n*S + j + k]
#pragma unroll
    for (int t = 0; t < 4; ++t)
      O[t] = wmma_bf16(pa, load_bT_frag(Vp, S, j, t * 16, lane), O[t]);
  }

  // normalize and merge heads: ctx[(b*S+q)*1024 + h*64 + col]
#pragma unroll
  for (int t = 0; t < 4; ++t) {
#pragma unroll
    for (int v = 0; v < 8; ++v) {
      const int qr = q0 + hi * 8 + v;
      const size_t row = (size_t)b * S + qr;
      ctx[row * 1024 + h * 64 + t * 16 + nl] = (__bf16)(O[t][v] / lrow[v]);
    }
  }
}

// ---------------------------------------------------------------------------
// y = LayerNorm(a + r) * g + beta over D=1024. One block (256 thr) per row.
// ---------------------------------------------------------------------------
__global__ void __launch_bounds__(256) k_add_ln(
    const float* __restrict__ a, const float* __restrict__ r,
    const float* __restrict__ g, const float* __restrict__ beta,
    float* __restrict__ outf, __bf16* __restrict__ outb) {
  __shared__ float red[256];
  const int row = blockIdx.x;
  const int t = threadIdx.x;
  const size_t base = (size_t)row * 1024;

  float x[4];
  float s = 0.0f;
#pragma unroll
  for (int i = 0; i < 4; ++i) {
    const int c = t + i * 256;
    x[i] = a[base + c] + r[base + c];
    s += x[i];
  }
  red[t] = s;
  __syncthreads();
  for (int off = 128; off > 0; off >>= 1) {
    if (t < off) red[t] += red[t + off];
    __syncthreads();
  }
  const float mu = red[0] * (1.0f / 1024.0f);
  __syncthreads();

  s = 0.0f;
#pragma unroll
  for (int i = 0; i < 4; ++i) { const float d = x[i] - mu; s += d * d; }
  red[t] = s;
  __syncthreads();
  for (int off = 128; off > 0; off >>= 1) {
    if (t < off) red[t] += red[t + off];
    __syncthreads();
  }
  const float rstd = rsqrtf(red[0] * (1.0f / 1024.0f) + 1e-5f);

#pragma unroll
  for (int i = 0; i < 4; ++i) {
    const int c = t + i * 256;
    const float y = (x[i] - mu) * rstd * g[c] + beta[c];
    if (outf) outf[base + c] = y;
    if (outb) outb[base + c] = (__bf16)y;
  }
}

// ---------------------------------------------------------------------------
// Host orchestration
// ---------------------------------------------------------------------------
extern "C" void kernel_launch(void* const* d_in, const int* in_sizes, int n_in,
                              void* d_out, int out_size, void* d_ws, size_t ws_size,
                              hipStream_t stream) {
  (void)in_sizes; (void)n_in; (void)out_size; (void)ws_size;
  constexpr int Bb = 2, S = 2048, D = 1024, H = 16, FF = 4096;
  constexpr int M = Bb * S;

  const float* src  = (const float*)d_in[0];
  const float* Wq   = (const float*)d_in[1];
  const float* bq   = (const float*)d_in[2];
  const float* Wk   = (const float*)d_in[3];
  const float* bk   = (const float*)d_in[4];
  const float* Wv   = (const float*)d_in[5];
  const float* bv   = (const float*)d_in[6];
  const float* Wo   = (const float*)d_in[7];
  const float* bo   = (const float*)d_in[8];
  const float* W1   = (const float*)d_in[9];
  const float* b1   = (const float*)d_in[10];
  const float* W2   = (const float*)d_in[11];
  const float* b2   = (const float*)d_in[12];
  const float* ln1g = (const float*)d_in[13];
  const float* ln1b = (const float*)d_in[14];
  const float* ln2g = (const float*)d_in[15];
  const float* ln2b = (const float*)d_in[16];

  char* w = (char*)d_ws;
  auto carve = [&](size_t bytes) -> char* {
    char* p = w;
    w += (bytes + 255) & ~(size_t)255;
    return p;
  };
  __bf16* src_bf = (__bf16*)carve((size_t)M * D * 2);
  __bf16* Wq_t   = (__bf16*)carve((size_t)D * D * 2);   // [N][K]
  __bf16* Wk_t   = (__bf16*)carve((size_t)D * D * 2);
  __bf16* Wv_t   = (__bf16*)carve((size_t)D * D * 2);
  __bf16* Wo_t   = (__bf16*)carve((size_t)D * D * 2);
  __bf16* W1_t   = (__bf16*)carve((size_t)D * FF * 2);  // [FF][D]
  __bf16* W2_t   = (__bf16*)carve((size_t)FF * D * 2);  // [D][FF]
  __bf16* Qh     = (__bf16*)carve((size_t)M * D * 2);   // [bh][s][64]
  __bf16* Kh     = (__bf16*)carve((size_t)M * D * 2);   // [bh][s][64]
  __bf16* Vt     = (__bf16*)carve((size_t)M * D * 2);   // [bh][d][s]
  __bf16* ctx_bf = (__bf16*)carve((size_t)M * D * 2);
  float*  tmpC   = (float*)carve((size_t)M * D * 4);
  float*  x_f    = (float*)carve((size_t)M * D * 4);
  __bf16* x_bf   = (__bf16*)carve((size_t)M * D * 2);
  __bf16* h1_bf  = (__bf16*)carve((size_t)M * FF * 2);

  k_f32_to_bf16<<<(M * D + 255) / 256, 256, 0, stream>>>(src, src_bf, M * D);
  k_f32_to_bf16_T<<<(D * D + 255) / 256, 256, 0, stream>>>(Wq, Wq_t, D, D);
  k_f32_to_bf16_T<<<(D * D + 255) / 256, 256, 0, stream>>>(Wk, Wk_t, D, D);
  k_f32_to_bf16_T<<<(D * D + 255) / 256, 256, 0, stream>>>(Wv, Wv_t, D, D);
  k_f32_to_bf16_T<<<(D * D + 255) / 256, 256, 0, stream>>>(Wo, Wo_t, D, D);
  k_f32_to_bf16_T<<<(D * FF + 255) / 256, 256, 0, stream>>>(W1, W1_t, D, FF);
  k_f32_to_bf16_T<<<(FF * D + 255) / 256, 256, 0, stream>>>(W2, W2_t, FF, D);

  const dim3 gDD(D / 64, M / 256);     // N=1024 GEMMs
  const dim3 gFF(FF / 64, M / 256);    // N=4096 GEMM
  const int eltBlocks = (M * D + 255) / 256;

  // Q = src@Wq+bq (scaled 1/8 into heads); K,V similarly (V transposed)
  k_gemm_bf16<<<gDD, 256, 0, stream>>>(src_bf, Wq_t, bq, tmpC, nullptr, M, D, D, 0);
  k_split_heads<<<eltBlocks, 256, 0, stream>>>(tmpC, Qh, 0.125f, M * D);
  k_gemm_bf16<<<gDD, 256, 0, stream>>>(src_bf, Wk_t, bk, tmpC, nullptr, M, D, D, 0);
  k_split_heads<<<eltBlocks, 256, 0, stream>>>(tmpC, Kh, 1.0f, M * D);
  k_gemm_bf16<<<gDD, 256, 0, stream>>>(src_bf, Wv_t, bv, tmpC, nullptr, M, D, D, 0);
  k_split_heads_T<<<eltBlocks, 256, 0, stream>>>(tmpC, Vt, M * D);

  // flash attention -> ctx (heads merged, bf16)
  k_flash_attn<<<dim3(S / 64, Bb * H), 128, 0, stream>>>(Qh, Kh, Vt, ctx_bf, S, H);

  // out projection + residual LN1
  k_gemm_bf16<<<gDD, 256, 0, stream>>>(ctx_bf, Wo_t, bo, tmpC, nullptr, M, D, D, 0);
  k_add_ln<<<M, 256, 0, stream>>>(src, tmpC, ln1g, ln1b, x_f, x_bf);

  // FFN: relu(x@W1+b1)@W2+b2, then residual LN2 -> d_out
  k_gemm_bf16<<<gFF, 256, 0, stream>>>(x_bf, W1_t, b1, nullptr, h1_bf, M, FF, D, 1);
  k_gemm_bf16<<<gDD, 256, 0, stream>>>(h1_bf, W2_t, b2, tmpC, nullptr, M, D, FF, 0);
  k_add_ln<<<M, 256, 0, stream>>>(x_f, tmpC, ln2g, ln2b, (float*)d_out, nullptr);
}
